// FastMambaBlock_84267258347691
// MI455X (gfx1250) — matrix operand order repaired and checked by reading, not compiled, not run on hardware
//
#include <hip/hip_runtime.h>
#include <hip/hip_bf16.h>

// ---------------- problem constants ----------------
#define T_LEN   1024
#define BATCH   16
#define D_MODEL 1024
#define D_INNER 2048
#define D_STATE 16
#define D_CONV  4
#define DT_RANK 64
#define TB      (T_LEN * BATCH)          // 16384 GEMM rows (multiple of 16)

typedef __attribute__((ext_vector_type(16))) _Float16 v16h;
typedef __attribute__((ext_vector_type(8)))  _Float16 v8h;
typedef __attribute__((ext_vector_type(8)))  float    v8f;

// ---------------- f32 -> f16 conversion ----------------
__global__ void cvt_f32_to_f16(_Float16* __restrict__ dst,
                               const float* __restrict__ src, size_t n) {
    size_t i = (size_t)blockIdx.x * blockDim.x + threadIdx.x;
    if (i < n) dst[i] = (_Float16)src[i];
}

// ---------------- generic WMMA GEMM: C[m,n] = sum_k A[m,k] * W[n,k] ------------
// One wave per (16*MT) x (16*NT) tile, f32 accumulate, no per-lane guards:
// grid must tile M and N exactly (all our N are multiples of 16*NT).
//
// Fragment layouts used here (16-bit WMMA, wave32; contiguous k order, same for A and B):
//  A (16x32 MxK): lane 0-15 = row M=lane, elems 0..15 = K[k0..k0+15];
//                 lanes 16-31 get +16 K offset.
//  B (32x16 KxN) supplied as W rows (K contiguous): lane holds column n0+lane,
//                 one contiguous 16-half chunk at K[k0 + (lane>>4)*16].
template <int MT, int NT, bool SIDE16>
__global__ void __launch_bounds__(32)
wmma_gemm(const _Float16* __restrict__ A, int lda,
          const _Float16* __restrict__ W, int ldb,
          float* __restrict__ C, int ldc, int K,
          _Float16* __restrict__ C16, int ldc16, int n16) {
    const int lane = threadIdx.x & 31;
    const int r  = lane & 15;
    const int hi = lane >> 4;
    const int m0 = blockIdx.y * (16 * MT);
    const int n0 = blockIdx.x * (16 * NT);

    // loop-invariant per-lane base pointers
    const _Float16* pa[MT];
#pragma unroll
    for (int mi = 0; mi < MT; ++mi)
        pa[mi] = A + (size_t)(m0 + mi * 16 + r) * lda + hi * 16;
    const _Float16* pb[NT];
#pragma unroll
    for (int nj = 0; nj < NT; ++nj)
        pb[nj] = W + (size_t)(n0 + nj * 16 + r) * ldb + hi * 16;

    v8f acc[MT][NT];
#pragma unroll
    for (int mi = 0; mi < MT; ++mi)
#pragma unroll
        for (int nj = 0; nj < NT; ++nj)
            acc[mi][nj] = (v8f){};

    for (int k0 = 0; k0 < K; k0 += 32) {
        if (k0 + 512 < K) {               // pull lines toward WGP (global_prefetch_b8)
#pragma unroll
            for (int mi = 0; mi < MT; ++mi) __builtin_prefetch(pa[mi] + k0 + 512, 0, 0);
#pragma unroll
            for (int nj = 0; nj < NT; ++nj) __builtin_prefetch(pb[nj] + k0 + 512, 0, 0);
        }
        v16h af[MT];
#pragma unroll
        for (int mi = 0; mi < MT; ++mi) {
            v8h lo = *(const v8h*)(pa[mi] + k0);
            v8h hb = *(const v8h*)(pa[mi] + k0 + 8);
#pragma unroll
            for (int i = 0; i < 8; ++i) { af[mi][i] = lo[i]; af[mi][i + 8] = hb[i]; }
        }
#pragma unroll
        for (int nj = 0; nj < NT; ++nj) {
            v16h bf = *(const v16h*)(pb[nj] + k0);
#pragma unroll
            for (int mi = 0; mi < MT; ++mi)
                acc[mi][nj] = __builtin_amdgcn_wmma_f32_16x16x32_f16(
                    false, af[mi], false, bf, (short)0, acc[mi][nj], false, false);
        }
    }

    // C/D layout: VGPR i, lanes 0-15 -> (M = mtile+i, N = lane);
    //             lanes 16-31 -> (M = mtile+8+i, N = lane-16)
    const int nc = lane & 15;
    const int mo = (lane >> 4) * 8;
#pragma unroll
    for (int mi = 0; mi < MT; ++mi) {
        const int mb = m0 + mi * 16 + mo;
#pragma unroll
        for (int nj = 0; nj < NT; ++nj) {
            const int n = n0 + nj * 16 + nc;
#pragma unroll
            for (int i = 0; i < 8; ++i) {
                float v = acc[mi][nj][i];
                C[(size_t)(mb + i) * ldc + n] = v;
                if (SIDE16 && n < n16)          // tile-uniform (n16 multiple of 16)
                    C16[(size_t)(mb + i) * ldc16 + n] = (_Float16)v;
            }
        }
    }
}

// ---------------- causal depthwise conv (width 4) with resets + SiLU -----------
// Parallel over (t, b, d). Also emits final conv state (taps at t = T-1).
__global__ void conv_silu_kernel(const float* __restrict__ XZ,       // (TB, 4096), xi = cols [0,2048)
                                 const int*   __restrict__ dones,    // (T, B)
                                 const float* __restrict__ conv_w,   // (2048, 4)
                                 const float* __restrict__ conv_b,   // (2048)
                                 const float* __restrict__ conv0,    // (B, 2048, 4)
                                 float*  __restrict__ XC,            // (TB, 2048)
                                 _Float16* __restrict__ XCh,         // (TB, 2048)
                                 float*  __restrict__ conv_out) {    // (B, 2048, 4)
    size_t idx = (size_t)blockIdx.x * blockDim.x + threadIdx.x;
    if (idx >= (size_t)TB * D_INNER) return;
    int d = (int)(idx % D_INNER);
    int b = (int)((idx / D_INNER) % BATCH);
    int t = (int)(idx / ((size_t)D_INNER * BATCH));

    float taps[4];
    float alive = 1.f;
#pragma unroll
    for (int j = 3; j >= 0; --j) {
        int s = t - 3 + j;                           // source time for tap j
        float v;
        if (s >= 0) v = XZ[((size_t)s * BATCH + b) * (2 * D_INNER) + d];
        else        v = conv0[((size_t)b * D_INNER + d) * D_CONV + (s + 4)];
        taps[j] = alive * v;
        // tap j-1 additionally requires no reset at time s (reset[s]=dones[s-1])
        if (s > 0 && dones[(size_t)(s - 1) * BATCH + b]) alive = 0.f;
    }
    float acc = conv_b[d];
#pragma unroll
    for (int j = 0; j < 4; ++j) acc += taps[j] * conv_w[(size_t)d * D_CONV + j];
    float xc = acc / (1.f + __expf(-acc));           // SiLU
    XC[idx]  = xc;
    XCh[idx] = (_Float16)xc;
    if (t == T_LEN - 1) {
#pragma unroll
        for (int j = 0; j < 4; ++j)
            conv_out[((size_t)b * D_INNER + d) * D_CONV + j] = taps[j];
    }
}

// ---------------- the only sequential part: SSM recurrence ---------------------
// 32768 independent channels (b,d); each thread keeps 16 states in registers.
// B/C (32 floats per (t,b)) staged in LDS, shared by all 256 d-threads of the block.
__global__ void __launch_bounds__(256)
scan_kernel(const float* __restrict__ XZ,      // (TB, 4096): z = cols [2048,4096)
            const float* __restrict__ XC,      // (TB, 2048)
            const float* __restrict__ XDB,     // (TB, 96): [dt_in(64) | B(16) | C(16)]
            const float* __restrict__ DTPRE,   // (TB, 2048) pre-bias dt
            const int*   __restrict__ dones,   // (T, B)
            const float* __restrict__ ssm0,    // (B, 2048, 16)
            const float* __restrict__ A_log,   // (2048, 16)
            const float* __restrict__ D_param, // (2048)
            const float* __restrict__ dt_bias, // (2048)
            _Float16* __restrict__ YH,         // (TB, 2048) gated y, f16 for out GEMM
            float* __restrict__ ssm_out) {     // (B, 2048, 16)
    const int b  = blockIdx.x >> 3;
    const int d  = ((blockIdx.x & 7) << 8) + threadIdx.x;
    __shared__ float sBC[32];
    __shared__ float sReset;

    float An[D_STATE], h[D_STATE];
#pragma unroll
    for (int n = 0; n < D_STATE; ++n) {
        An[n] = -__expf(A_log[(size_t)d * D_STATE + n]);
        h[n]  = ssm0[((size_t)b * D_INNER + d) * D_STATE + n];
    }
    const float Dp = D_param[d];
    const float bias = dt_bias[d];

    for (int t = 0; t < T_LEN; ++t) {
        const size_t row = (size_t)t * BATCH + b;
        if (threadIdx.x < 32) sBC[threadIdx.x] = XDB[row * 96 + DT_RANK + threadIdx.x];
        if (threadIdx.x == 32)
            sReset = (t > 0) ? (float)dones[(size_t)(t - 1) * BATCH + b] : 0.f;
        __syncthreads();

        const float m   = 1.f - sReset;
        float dtp = DTPRE[row * D_INNER + d] + bias;
        float dt  = (dtp > 20.f) ? dtp : __logf(1.f + __expf(dtp));   // softplus
        const float xc   = XC[row * D_INNER + d];
        const float dtxc = dt * xc;
        float y = 0.f;
#pragma unroll
        for (int n = 0; n < D_STATE; ++n) {
            float hv = h[n] * m;
            hv = hv * __expf(dt * An[n]) + dtxc * sBC[n];
            y += hv * sBC[16 + n];
            h[n] = hv;
        }
        y += Dp * xc;
        const float z = XZ[row * (2 * D_INNER) + D_INNER + d];
        y *= z / (1.f + __expf(-z));                                   // * silu(z)
        YH[row * D_INNER + d] = (_Float16)y;
        __syncthreads();
    }
#pragma unroll
    for (int n = 0; n < D_STATE; ++n)
        ssm_out[((size_t)b * D_INNER + d) * D_STATE + n] = h[n];
}

// ---------------- host launch ---------------------------------------------------
static inline size_t cdiv(size_t a, size_t b) { return (a + b - 1) / b; }

extern "C" void kernel_launch(void* const* d_in, const int* in_sizes, int n_in,
                              void* d_out, int out_size, void* d_ws, size_t ws_size,
                              hipStream_t stream) {
    const float* x_seq     = (const float*)d_in[0];
    const float* ssm0      = (const float*)d_in[1];
    const float* conv0     = (const float*)d_in[2];
    const int*   dones     = (const int*)  d_in[3];
    const float* in_proj_w = (const float*)d_in[4];
    const float* conv_w    = (const float*)d_in[5];
    const float* conv_b    = (const float*)d_in[6];
    const float* x_proj_w  = (const float*)d_in[7];
    const float* dt_proj_w = (const float*)d_in[8];
    const float* dt_proj_b = (const float*)d_in[9];
    const float* A_log     = (const float*)d_in[10];
    const float* D_param   = (const float*)d_in[11];
    const float* out_proj_w= (const float*)d_in[12];

    float* out      = (float*)d_out;                         // (T,B,1024)
    float* ssm_out  = out + (size_t)TB * D_MODEL;            // (B,2048,16)
    float* conv_out = ssm_out + (size_t)BATCH * D_INNER * D_STATE; // (B,2048,4)

    // workspace carve-out (256B aligned)
    char* ws = (char*)d_ws;
    size_t off = 0;
    auto take = [&](size_t bytes) -> char* {
        char* p = ws + off;
        off = (off + bytes + 255) & ~(size_t)255;
        return p;
    };
    _Float16* W1h   = (_Float16*)take((size_t)2 * D_INNER * D_MODEL * 2); // (4096,1024)
    _Float16* Wouth = (_Float16*)take((size_t)D_MODEL * D_INNER * 2);     // (1024,2048)
    _Float16* Wxph  = (_Float16*)take((size_t)96 * D_INNER * 2);          // (96,2048)
    _Float16* Wdth  = (_Float16*)take((size_t)D_INNER * DT_RANK * 2);     // (2048,64)
    _Float16* Xh    = (_Float16*)take((size_t)TB * D_MODEL * 2);          // (16384,1024)
    float*    XZ    = (float*)   take((size_t)TB * 2 * D_INNER * 4);      // (16384,4096)
    float*    XC    = (float*)   take((size_t)TB * D_INNER * 4);          // (16384,2048)
    _Float16* XCh   = (_Float16*)take((size_t)TB * D_INNER * 2);
    float*    XDB   = (float*)   take((size_t)TB * 96 * 4);
    _Float16* DT16  = (_Float16*)take((size_t)TB * DT_RANK * 2);
    float*    DTPRE = (float*)   take((size_t)TB * D_INNER * 4);
    _Float16* YH    = (_Float16*)take((size_t)TB * D_INNER * 2);
    (void)ws_size; (void)in_sizes; (void)n_in; (void)out_size;

    // 1) f16 conversions (weights stay L2-resident; reused 1024x)
    {
        auto launch_cvt = [&](_Float16* dst, const float* src, size_t n) {
            cvt_f32_to_f16<<<dim3((unsigned)cdiv(n, 256)), 256, 0, stream>>>(dst, src, n);
        };
        launch_cvt(Xh,    x_seq,      (size_t)TB * D_MODEL);
        launch_cvt(W1h,   in_proj_w,  (size_t)2 * D_INNER * D_MODEL);
        launch_cvt(Wxph,  x_proj_w,   (size_t)96 * D_INNER);
        launch_cvt(Wdth,  dt_proj_w,  (size_t)D_INNER * DT_RANK);
        launch_cvt(Wouth, out_proj_w, (size_t)D_MODEL * D_INNER);
    }

    // 2) in_proj GEMM: XZ = X @ W1^T   (M=16384, N=4096, K=1024) — parallel over time
    wmma_gemm<2, 4, false><<<dim3(4096 / 64, TB / 32), 32, 0, stream>>>(
        Xh, D_MODEL, W1h, D_MODEL, XZ, 2 * D_INNER, D_MODEL, nullptr, 0, 0);

    // 3) causal conv + SiLU (parallel over t,b,d), emits final conv state
    conv_silu_kernel<<<dim3((unsigned)cdiv((size_t)TB * D_INNER, 256)), 256, 0, stream>>>(
        XZ, dones, conv_w, conv_b, conv0, XC, XCh, conv_out);

    // 4) x_proj GEMM: XDB = xc @ Wxp^T (N=96 = 2 blocks x 3 tiles, K=2048);
    //    f16 side-store of the DT_RANK columns for the next GEMM
    wmma_gemm<2, 3, true><<<dim3(96 / 48, TB / 32), 32, 0, stream>>>(
        XCh, D_INNER, Wxph, D_INNER, XDB, 96, D_INNER, DT16, DT_RANK, DT_RANK);

    // 5) dt_proj GEMM: DTPRE = dt_in @ Wdt^T (N=2048, K=64)
    wmma_gemm<2, 4, false><<<dim3(2048 / 64, TB / 32), 32, 0, stream>>>(
        DT16, DT_RANK, Wdth, DT_RANK, DTPRE, D_INNER, DT_RANK, nullptr, 0, 0);

    // 6) sequential SSM scan: 32768 independent channels, states in registers
    scan_kernel<<<dim3(BATCH * (D_INNER / 256)), 256, 0, stream>>>(
        XZ, XC, XDB, DTPRE, dones, ssm0, A_log, D_param, dt_proj_b, YH, ssm_out);

    // 7) out_proj GEMM: out = y @ Wout^T (N=1024, K=2048) -> straight into d_out
    wmma_gemm<2, 4, false><<<dim3(D_MODEL / 64, TB / 32), 32, 0, stream>>>(
        YH, D_INNER, Wouth, D_INNER, out, D_MODEL, D_INNER, nullptr, 0, 0);
}
